// Qwen3VLMoeTextSparseMoeBlock_11716670783688
// MI455X (gfx1250) — compile-verified, matrix operand order checked by reading, not run on hardware
//
#include <hip/hip_runtime.h>

#define HIDDEN 2048
#define NEXP   64
#define INTER  768
#define TWOI   1536
#define TOPK   8
#define NTOK   2048   // B*S = 2*1024
#define MTILE  16

typedef __attribute__((ext_vector_type(16))) __bf16 v16bf;
typedef __attribute__((ext_vector_type(8)))  float  v8f;
typedef __attribute__((ext_vector_type(4)))  unsigned int v4u;
typedef __attribute__((ext_vector_type(8)))  int v8i;
typedef __attribute__((ext_vector_type(4)))  int v4i;

static __device__ __forceinline__ __bf16 to_bf16(float f) { return (__bf16)f; }

// ---------------------------------------------------------------- zero init
__global__ __launch_bounds__(256) void k_zero(float* __restrict__ out0, int n4,
                                              int* __restrict__ counts) {
  int i = blockIdx.x * blockDim.x + threadIdx.x;
  if (i < n4) ((float4*)out0)[i] = float4{0.f, 0.f, 0.f, 0.f};
  if (blockIdx.x == 0 && threadIdx.x < NEXP) counts[threadIdx.x] = 0;
}

// ---------------------------------------------------------------- router
__global__ __launch_bounds__(64) void k_router(const float* __restrict__ x,
                                               const float* __restrict__ wr,
                                               float* __restrict__ logits_out,
                                               int*   __restrict__ topk_idx,
                                               float* __restrict__ topk_w) {
  __shared__ float xs[HIDDEN];
  __shared__ float lg[NEXP];
  const int t   = blockIdx.x;
  const int tid = threadIdx.x;

  const float4* xr4 = (const float4*)(x + (size_t)t * HIDDEN);
  float4* xs4 = (float4*)xs;
  for (int i = tid; i < HIDDEN / 4; i += 64) xs4[i] = xr4[i];
  __syncthreads();

  const float4* w4 = (const float4*)(wr + (size_t)tid * HIDDEN);
  float acc = 0.f;
  for (int i = 0; i < HIDDEN / 4; ++i) {
    float4 a = xs4[i];
    float4 b = w4[i];
    acc += a.x * b.x + a.y * b.y + a.z * b.z + a.w * b.w;
  }
  lg[tid] = acc;
  logits_out[(size_t)t * NEXP + tid] = acc;   // raw logits (reference output #2)
  __syncthreads();

  if (tid == 0) {
    int   sel[TOPK];
    float sv[TOPK];
    unsigned long long used = 0ull;
    for (int k = 0; k < TOPK; ++k) {
      float best = -3.4e38f; int bi = 0;
      for (int e = 0; e < NEXP; ++e)
        if (!((used >> e) & 1ull) && lg[e] > best) { best = lg[e]; bi = e; }
      used |= 1ull << bi; sel[k] = bi; sv[k] = best;
    }
    // normalized top-k weights == softmax over the selected logits
    float m = sv[0], s = 0.f, ew[TOPK];
    for (int k = 0; k < TOPK; ++k) { ew[k] = __expf(sv[k] - m); s += ew[k]; }
    float inv = 1.f / s;
    for (int k = 0; k < TOPK; ++k) {
      topk_idx[t * TOPK + k] = sel[k];
      topk_w [t * TOPK + k] = ew[k] * inv;
    }
  }
}

// ---------------------------------------------------------------- list build
__global__ __launch_bounds__(64) void k_build(const int*   __restrict__ topk_idx,
                                              const float* __restrict__ topk_w,
                                              int*   __restrict__ counts,
                                              int*   __restrict__ lists,
                                              float* __restrict__ lwts) {
  const int e = threadIdx.x;
  int c = 0;
  for (int t = 0; t < NTOK; ++t) {
    for (int k = 0; k < TOPK; ++k) {
      if (topk_idx[t * TOPK + k] == e) {
        lists[e * NTOK + c] = t;
        lwts [e * NTOK + c] = topk_w[t * TOPK + k];
        ++c;
      }
    }
  }
  counts[e] = c;
}

// ---------------------------------------------------------------- fused expert
// block = 256 threads (8 waves), tile = 16 tokens of one expert.
// X tile staged by the Tensor Data Mover in GATHER mode (16 token-row indices,
// 256 f32 per row per chunk), double-buffered in LDS so the next TDM load
// overlaps WMMA on the current chunk.
__global__ __launch_bounds__(256) void k_expert(const float* __restrict__ x,
                                                const float* __restrict__ gup,
                                                const float* __restrict__ dwn,
                                                const int*   __restrict__ counts,
                                                const int*   __restrict__ lists,
                                                const float* __restrict__ lwts,
                                                float* __restrict__ out) {
  const int e    = blockIdx.y;
  const int tile = blockIdx.x;
  const int cnt  = counts[e];
  if (tile * MTILE >= cnt) return;

  __shared__ float  XsF[2][MTILE][256];  // 32 KB : double-buffered TDM gather dst
  __shared__ __bf16 Hs[MTILE][INTER];    // 24 KB : silu(gate)*up activations
  __shared__ int    toks[MTILE];
  __shared__ float  twt [MTILE];

  const int tid  = threadIdx.x;
  const int lane = tid & 31;
  const int wv   = tid >> 5;                 // wave id 0..7
  const int l16  = lane & 15;                // A-row / C-col within tile
  const int klo  = (lane < 16) ? 0 : 8;      // K-half select (ISA 16-bit A layout)
  const int rbase = (lane < 16) ? 0 : 8;     // C/D row base (ISA f32 C layout)

  if (tid < MTILE) {
    int idx = tile * MTILE + tid;
    bool v  = idx < cnt;
    toks[tid] = v ? lists[e * NTOK + idx] : 0;
    twt [tid] = v ? lwts [e * NTOK + idx] : 0.f;
  }
  __syncthreads();

  // ---- Tensor DMA descriptor (gather mode, 16-bit indices, f32 elements) ----
  // group1: data_size=4B; tensor_dim0=2048, tensor_dim1=2048 (OOB bounds);
  //         tile_dim0=256 (chunk width), tile_dim1=16 (#valid row indices);
  //         tensor_dim0_stride=2048 elements.
  const v8i g1 = { 0x00020000, 0x08000000, 0x08000000, 0x01000000,
                   16, 2048, 0, 0 };
  // groups 2/3: 16 packed 16-bit gathered row (token) indices
  const int i0 = __builtin_amdgcn_readfirstlane(((toks[1] & 0xffff) << 16) | (toks[0] & 0xffff));
  const int i1 = __builtin_amdgcn_readfirstlane(((toks[3] & 0xffff) << 16) | (toks[2] & 0xffff));
  const int i2 = __builtin_amdgcn_readfirstlane(((toks[5] & 0xffff) << 16) | (toks[4] & 0xffff));
  const int i3 = __builtin_amdgcn_readfirstlane(((toks[7] & 0xffff) << 16) | (toks[6] & 0xffff));
  const int i4 = __builtin_amdgcn_readfirstlane(((toks[9]  & 0xffff) << 16) | (toks[8]  & 0xffff));
  const int i5 = __builtin_amdgcn_readfirstlane(((toks[11] & 0xffff) << 16) | (toks[10] & 0xffff));
  const int i6 = __builtin_amdgcn_readfirstlane(((toks[13] & 0xffff) << 16) | (toks[12] & 0xffff));
  const int i7 = __builtin_amdgcn_readfirstlane(((toks[15] & 0xffff) << 16) | (toks[14] & 0xffff));
  const v4i g2  = { i0, i1, i2, i3 };
  const v4i g3  = { i4, i5, i6, i7 };
  const v8i g23 = { i0, i1, i2, i3, i4, i5, i6, i7 };   // 6-arg form extra group
  const unsigned lds0 = (unsigned)(uintptr_t)&XsF[0][0][0];
  const unsigned lds1 = (unsigned)(uintptr_t)&XsF[1][0][0];
  const unsigned long long xbase = (unsigned long long)(uintptr_t)x;

  auto tdm_issue = [&](int kc, int buf) {
    unsigned long long ga = xbase + (unsigned long long)kc * 4ull;
    // group0: count=1 | gather_mode(bit31) | 16-bit idx; lds addr; 57-bit
    // global addr; type=2 ("image") in bits [127:126]
    v4u g0 = { 0x80000001u,
               buf ? lds1 : lds0,
               (unsigned)ga,
               (unsigned)((ga >> 32) & 0x01FFFFFFu) | 0x80000000u };
    __builtin_amdgcn_tensor_load_to_lds(g0, g1, g2, g3, g23, 0);
  };

  if (wv == 0) tdm_issue(0, 0);          // prime the pipeline (chunk 0)

  const float* Wg = gup + (size_t)e * HIDDEN * TWOI;

  // GEMM1 accumulators: per wave, 6 gate tiles + 6 up tiles (96 paired columns)
  v8f accG[6], accU[6];
#pragma unroll
  for (int j = 0; j < 6; ++j)
#pragma unroll
    for (int v = 0; v < 8; ++v) { accG[j][v] = 0.f; accU[j][v] = 0.f; }

  for (int it = 0; it < HIDDEN / 256; ++it) {
    if (wv == 0) __builtin_amdgcn_s_wait_tensorcnt(0);  // chunk `it` landed
    __syncthreads();                                    // publish to all waves
    if (wv == 0 && it + 1 < HIDDEN / 256)
      tdm_issue((it + 1) * 256, (it + 1) & 1);          // overlap next chunk
    const float* Xb = &XsF[it & 1][0][0];
    const int kc = it * 256;
    if (kc + 256 < HIDDEN)                              // weights toward L2
      __builtin_prefetch(Wg + (size_t)(kc + 256) * TWOI + wv * 96, 0, 1);

    for (int ks = 0; ks < 256; ks += 32) {
      v16bf a;
#pragma unroll
      for (int i = 0; i < 8; ++i) {
        a[i]     = to_bf16(Xb[l16 * 256 + ks + klo + i]);
        a[i + 8] = to_bf16(Xb[l16 * 256 + ks + 16 + klo + i]);
      }
      const int kg = kc + ks;
#pragma unroll
      for (int j = 0; j < 6; ++j) {
        const int cg = wv * 96 + j * 16 + l16;          // gate column
        const int cu = INTER + cg;                      // paired up column
        v16bf bg, bu;
#pragma unroll
        for (int i = 0; i < 8; ++i) {
          bg[i]     = to_bf16(Wg[(size_t)(kg + klo + i) * TWOI + cg]);
          bg[i + 8] = to_bf16(Wg[(size_t)(kg + 16 + klo + i) * TWOI + cg]);
          bu[i]     = to_bf16(Wg[(size_t)(kg + klo + i) * TWOI + cu]);
          bu[i + 8] = to_bf16(Wg[(size_t)(kg + 16 + klo + i) * TWOI + cu]);
        }
        accG[j] = __builtin_amdgcn_wmma_f32_16x16x32_bf16(
            false, a, false, bg, (short)0, accG[j], false, false);
        accU[j] = __builtin_amdgcn_wmma_f32_16x16x32_bf16(
            false, a, false, bu, (short)0, accU[j], false, false);
      }
    }
  }

  // h = up * silu(gate), store bf16 into LDS
  __syncthreads();
#pragma unroll
  for (int j = 0; j < 6; ++j)
#pragma unroll
    for (int v = 0; v < 8; ++v) {
      float g = accG[j][v], u = accU[j][v];
      float h = u * (g / (1.f + __expf(-g)));
      Hs[rbase + v][wv * 96 + j * 16 + l16] = to_bf16(h);
    }
  __syncthreads();

  // GEMM2: h(16 x 768) @ Wd(768 x 2048); each wave owns 256 output columns
  const float* Wd = dwn + (size_t)e * INTER * HIDDEN;
  v8f acc2[16];
#pragma unroll
  for (int j = 0; j < 16; ++j)
#pragma unroll
    for (int v = 0; v < 8; ++v) acc2[j][v] = 0.f;

  for (int kg = 0; kg < INTER; kg += 32) {
    v16bf a;
#pragma unroll
    for (int i = 0; i < 8; ++i) {
      a[i]     = Hs[l16][kg + klo + i];
      a[i + 8] = Hs[l16][kg + 16 + klo + i];
    }
#pragma unroll
    for (int j = 0; j < 16; ++j) {
      const int nc = wv * 256 + j * 16 + l16;
      v16bf b;
#pragma unroll
      for (int i = 0; i < 8; ++i) {
        b[i]     = to_bf16(Wd[(size_t)(kg + klo + i) * HIDDEN + nc]);
        b[i + 8] = to_bf16(Wd[(size_t)(kg + 16 + klo + i) * HIDDEN + nc]);
      }
      acc2[j] = __builtin_amdgcn_wmma_f32_16x16x32_bf16(
          false, a, false, b, (short)0, acc2[j], false, false);
    }
  }

  // weighted combine into routed output
#pragma unroll
  for (int j = 0; j < 16; ++j) {
    const int nc = wv * 256 + j * 16 + l16;
#pragma unroll
    for (int v = 0; v < 8; ++v) {
      const int r = rbase + v;
      float wt = twt[r];
      if (wt != 0.f)
        atomicAdd(&out[(size_t)toks[r] * HIDDEN + nc], acc2[j][v] * wt);
    }
  }
}

// ---------------------------------------------------------------- launch
extern "C" void kernel_launch(void* const* d_in, const int* in_sizes, int n_in,
                              void* d_out, int out_size, void* d_ws, size_t ws_size,
                              hipStream_t stream) {
  const float* x   = (const float*)d_in[0];   // (2,1024,2048)
  const float* wr  = (const float*)d_in[1];   // (64,2048)
  const float* gup = (const float*)d_in[2];   // (64,2048,1536)
  const float* dwn = (const float*)d_in[3];   // (64,768,2048)

  float* out_routed = (float*)d_out;                          // T*H
  float* out_logits = out_routed + (size_t)NTOK * HIDDEN;     // T*E

  char* ws = (char*)d_ws;
  int*   counts   = (int*)ws;                                         // 64
  int*   topk_idx = (int*)(ws + 256);                                 // T*8
  float* topk_w   = (float*)(ws + 256 + (size_t)NTOK * TOPK * 4);     // T*8
  int*   lists    = (int*)(ws + 256 + (size_t)2 * NTOK * TOPK * 4);   // E*T
  float* lwts     = (float*)(ws + 256 + (size_t)2 * NTOK * TOPK * 4
                                      + (size_t)NEXP * NTOK * 4);     // E*T

  k_zero<<<(NTOK * HIDDEN / 4 + 255) / 256, 256, 0, stream>>>(
      out_routed, NTOK * HIDDEN / 4, counts);
  k_router<<<NTOK, 64, 0, stream>>>(x, wr, out_logits, topk_idx, topk_w);
  k_build<<<1, 64, 0, stream>>>(topk_idx, topk_w, counts, lists, lwts);
  dim3 grid(NTOK / MTILE, NEXP);
  k_expert<<<grid, 256, 0, stream>>>(x, gup, dwn, counts, lists, lwts, out_routed);
}
